// EPROP3_LSTM_67156108640260
// MI455X (gfx1250) — matrix operand order repaired and checked by reading, not compiled, not run on hardware
//
#include <hip/hip_runtime.h>

typedef _Float16 half_t;
typedef __attribute__((ext_vector_type(16))) _Float16 v16h;
typedef __attribute__((ext_vector_type(8)))  float    v8f;

#define BB 64
#define TT 128
#define II 256
#define HH 512
#define OO 128
#define GG (4*HH)
#define LPAD 8   // LDS row padding (halves) -> 4-bank stride, conflict-free frag loads

// ---------------------------------------------------------------------------
// helpers
// ---------------------------------------------------------------------------
__global__ void cvt_f32_f16_k(const float* __restrict__ in, half_t* __restrict__ out, int n) {
    int i = blockIdx.x * blockDim.x + threadIdx.x;
    if (i < n) out[i] = (half_t)in[i];
}

__global__ void zero_f32_k(float* __restrict__ p, int n) {
    int i = blockIdx.x * blockDim.x + threadIdx.x;
    if (i < n) p[i] = 0.0f;
}

// ---------------------------------------------------------------------------
// Generic WMMA GEMM:  C[M,N] = A[M,K] @ W[N,K]^T  (+ bias) (relu?)
//   One wave32 computes a 16x64 C strip; A frag loaded/converted once per
//   K-step, reused across 4 v_wmma_f32_16x16x32_f16 (independent accs).
// ---------------------------------------------------------------------------
template<bool A_IS_HALF, bool RELU, bool BIAS, bool OUT_HALF>
__global__ __launch_bounds__(256)
void wmma_gemm_k(const void* __restrict__ Av, int lda,
                 const half_t* __restrict__ W, int K,
                 const float* __restrict__ bias,
                 void* __restrict__ Cv, int ldc,
                 int Mtiles, int Ngroups /* N/64 */)
{
    const int lane = threadIdx.x & 31;
    const int wave = threadIdx.x >> 5;
    const int tile = blockIdx.x * (blockDim.x >> 5) + wave;
    if (tile >= Mtiles * Ngroups) return;          // uniform per wave, EXEC stays full

    const int tm = tile / Ngroups;
    const int tg = tile - tm * Ngroups;
    const int m0 = tm << 4;
    const int n0 = tg << 6;

    const int hi = lane >> 4;
    const int lm = lane & 15;
    const int arow = m0 + lm;

    const float*  Af = (const float*)Av;
    const half_t* Ah = (const half_t*)Av;

    const half_t* wp[4];
    #pragma unroll
    for (int s = 0; s < 4; ++s)
        wp[s] = W + (size_t)(n0 + s * 16 + lm) * K;

    v8f acc[4];
    #pragma unroll
    for (int s = 0; s < 4; ++s)
        #pragma unroll
        for (int r = 0; r < 8; ++r) acc[s][r] = 0.0f;

    const int kbA = hi * 8;
    const int kbB = hi * 16;

    for (int kk = 0; kk < K; kk += 32) {
        v16h a;
        #pragma unroll
        for (int j = 0; j < 8; ++j) {
            const int kA = kk + kbA + ((j < 4) ? (2 * j) : (8 + 2 * j));
            if (A_IS_HALF) {
                a[2*j]   = Ah[(size_t)arow * lda + kA];
                a[2*j+1] = Ah[(size_t)arow * lda + kA + 1];
            } else {
                a[2*j]   = (half_t)Af[(size_t)arow * lda + kA];
                a[2*j+1] = (half_t)Af[(size_t)arow * lda + kA + 1];
            }
        }
        #pragma unroll
        for (int s = 0; s < 4; ++s) {
            __builtin_prefetch(wp[s] + kk + 64, 0, 0);
            v16h b;
            #pragma unroll
            for (int j = 0; j < 8; ++j) {
                const int kB = kk + kbA + ((j < 4) ? (2 * j) : (8 + 2 * j));
                b[2*j]   = wp[s][kB];
                b[2*j+1] = wp[s][kB + 1];
            }
            acc[s] = __builtin_amdgcn_wmma_f32_16x16x32_f16(
                         false, a, false, b, (short)0, acc[s], false, false);
        }
    }

    #pragma unroll
    for (int s = 0; s < 4; ++s) {
        const float bval = BIAS ? bias[n0 + s * 16 + lm] : 0.0f;
        #pragma unroll
        for (int r = 0; r < 8; ++r) {
            float v = acc[s][r] + bval;
            if (RELU) v = v > 0.0f ? v : 0.0f;
            const size_t idx = (size_t)(m0 + r + hi * 8) * ldc + n0 + s * 16 + lm;
            if (OUT_HALF) ((half_t*)Cv)[idx] = (half_t)v;
            else          ((float*)Cv)[idx]  = v;
        }
    }
}

// ---------------------------------------------------------------------------
// Persistent fused LSTM recurrence.
//   grid = 4 workgroups (16 batch rows each), 512 threads = 16 waves.
//   Wave w owns h-columns [32w, 32w+32) *in each gate block*, so its 8
//   accumulator tiles hold z_i/z_f/z_g/z_o for the same (row,col) in the
//   same lane -> gates + c/h update are lane-local register math.
//   h double-buffered in LDS; c lives in registers; one barrier per step.
// ---------------------------------------------------------------------------
__global__ __launch_bounds__(512)
void lstm_persistent_k(const float* __restrict__ x,      // (B,T,I) fp32
                       const half_t* __restrict__ Zall,  // (B,T,4H) f16 or null
                       const half_t* __restrict__ Wih_h, // (4H,I) f16
                       const half_t* __restrict__ Whh_h, // (4H,H) f16
                       const float* __restrict__ b_lstm, // (4H)
                       half_t* __restrict__ hs,          // (B,T,H) f16 out
                       int useZall)
{
    __shared__ half_t hbuf[2][16][HH + LPAD];

    const int tid  = threadIdx.x;
    const int lane = tid & 31;
    const int wave = tid >> 5;          // 0..15
    const int hi   = lane >> 4;
    const int lm   = lane & 15;
    const int b0   = blockIdx.x * 16;   // batch-row base
    const int colw = wave * 32;         // this wave's h-column base

    // h_{-1} = 0
    for (int i = tid; i < 16 * (HH + LPAD); i += 512)
        ((half_t*)hbuf[0])[i] = (half_t)0.0f;
    __syncthreads();

    // cell state in registers: c[row = r+8*hi][col = colw+16s+lm]
    float creg[2][8];
    #pragma unroll
    for (int s = 0; s < 2; ++s)
        #pragma unroll
        for (int r = 0; r < 8; ++r) creg[s][r] = 0.0f;

    // per-lane bias and weight-row pointers for the 8 owned tiles [gate][s]
    float breg[4][2];
    const half_t* wph[4][2];
    const half_t* wpx[4][2];
    #pragma unroll
    for (int g = 0; g < 4; ++g) {
        #pragma unroll
        for (int s = 0; s < 2; ++s) {
            const int n = g * HH + colw + s * 16 + lm;
            breg[g][s] = b_lstm[n];
            wph[g][s]  = Whh_h + (size_t)n * HH;
            wpx[g][s]  = Wih_h + (size_t)n * II;
        }
    }

    const int kbA = hi * 8;
    const int kbB = hi * 16;

    int pb = 0;
    for (int t = 0; t < TT; ++t) {
        v8f acc[4][2];

        if (useZall) {
            // init from precomputed x-part (bias already folded in)
            #pragma unroll
            for (int g = 0; g < 4; ++g)
                #pragma unroll
                for (int s = 0; s < 2; ++s)
                    #pragma unroll
                    for (int r = 0; r < 8; ++r) {
                        const int m = r + hi * 8;
                        acc[g][s][r] = (float)Zall[((size_t)(b0 + m) * TT + t) * GG
                                                   + g * HH + colw + s * 16 + lm];
                    }
        } else {
            #pragma unroll
            for (int g = 0; g < 4; ++g)
                #pragma unroll
                for (int s = 0; s < 2; ++s)
                    #pragma unroll
                    for (int r = 0; r < 8; ++r) acc[g][s][r] = breg[g][s];
            // x-part GEMM in-loop (K = I)
            for (int kk = 0; kk < II; kk += 32) {
                v16h a;
                #pragma unroll
                for (int j = 0; j < 8; ++j) {
                    const int kA = kk + kbA + ((j < 4) ? (2 * j) : (8 + 2 * j));
                    const float* ap = x + ((size_t)(b0 + lm) * TT + t) * II + kA;
                    a[2*j]   = (half_t)ap[0];
                    a[2*j+1] = (half_t)ap[1];
                }
                #pragma unroll
                for (int g = 0; g < 4; ++g)
                    #pragma unroll
                    for (int s = 0; s < 2; ++s) {
                        v16h b;
                        #pragma unroll
                        for (int j = 0; j < 8; ++j) {
                            const int kB = kk + kbA + ((j < 4) ? (2 * j) : (8 + 2 * j));
                            b[2*j]   = wpx[g][s][kB];
                            b[2*j+1] = wpx[g][s][kB + 1];
                        }
                        acc[g][s] = __builtin_amdgcn_wmma_f32_16x16x32_f16(
                                        false, a, false, b, (short)0, acc[g][s], false, false);
                    }
            }
        }

        // h-part GEMM (K = H), A fragments from LDS (conflict-free via LPAD)
        for (int kk = 0; kk < HH; kk += 32) {
            v16h a;
            #pragma unroll
            for (int j = 0; j < 8; ++j) {
                const int kA = kk + kbA + ((j < 4) ? (2 * j) : (8 + 2 * j));
                a[2*j]   = hbuf[pb][lm][kA];
                a[2*j+1] = hbuf[pb][lm][kA + 1];
            }
            #pragma unroll
            for (int g = 0; g < 4; ++g)
                #pragma unroll
                for (int s = 0; s < 2; ++s) {
                    __builtin_prefetch(wph[g][s] + kk + 64, 0, 0);
                    v16h b;
                    #pragma unroll
                    for (int j = 0; j < 8; ++j) {
                        const int kB = kk + kbA + ((j < 4) ? (2 * j) : (8 + 2 * j));
                        b[2*j]   = wph[g][s][kB];
                        b[2*j+1] = wph[g][s][kB + 1];
                    }
                    acc[g][s] = __builtin_amdgcn_wmma_f32_16x16x32_f16(
                                    false, a, false, b, (short)0, acc[g][s], false, false);
                }
        }

        // gates + state update (lane-local), write h to LDS (next step) + global hs
        #pragma unroll
        for (int s = 0; s < 2; ++s) {
            #pragma unroll
            for (int r = 0; r < 8; ++r) {
                const float zi = acc[0][s][r];
                const float zf = acc[1][s][r];
                const float zg = acc[2][s][r];
                const float zo = acc[3][s][r];
                const float ig = 1.0f / (1.0f + __expf(-zi));
                const float fg = 1.0f / (1.0f + __expf(-zf));
                const float gg = tanhf(zg);
                const float og = 1.0f / (1.0f + __expf(-zo));
                const float cn = fg * creg[s][r] + ig * gg;
                creg[s][r] = cn;
                const float hn = og * tanhf(cn);
                const int m   = r + hi * 8;
                const int col = colw + s * 16 + lm;
                hbuf[pb ^ 1][m][col] = (half_t)hn;
                hs[((size_t)(b0 + m) * TT + t) * HH + col] = (half_t)hn;
            }
        }
        __syncthreads();
        pb ^= 1;
    }
}

// ---------------------------------------------------------------------------
// Host-side launch
// ---------------------------------------------------------------------------
static inline size_t wsAlloc(size_t& off, size_t bytes) {
    size_t o = off;
    off = (off + bytes + 255) & ~(size_t)255;
    return o;
}

extern "C" void kernel_launch(void* const* d_in, const int* in_sizes, int n_in,
                              void* d_out, int out_size, void* d_ws, size_t ws_size,
                              hipStream_t stream) {
    (void)in_sizes; (void)n_in; (void)out_size;

    const float* x      = (const float*)d_in[0];
    const float* Wih    = (const float*)d_in[1];
    const float* Whh    = (const float*)d_in[2];
    const float* b_lstm = (const float*)d_in[3];
    const float* Wd     = (const float*)d_in[4];
    const float* bd     = (const float*)d_in[5];
    const float* W1     = (const float*)d_in[6];
    const float* b1     = (const float*)d_in[7];
    const float* W2     = (const float*)d_in[8];
    const float* b2     = (const float*)d_in[9];
    const float* W3     = (const float*)d_in[10];
    const float* b3     = (const float*)d_in[11];
    const float* W4     = (const float*)d_in[12];
    const float* b4     = (const float*)d_in[13];

    char* ws = (char*)d_ws;
    size_t off = 0;
    half_t* Wih_h = (half_t*)(ws + wsAlloc(off, (size_t)GG * II * 2));
    half_t* Whh_h = (half_t*)(ws + wsAlloc(off, (size_t)GG * HH * 2));
    half_t* Wd_h  = (half_t*)(ws + wsAlloc(off, (size_t)OO * HH * 2));
    half_t* W1_h  = (half_t*)(ws + wsAlloc(off, (size_t)512 * OO * 2));
    half_t* W2_h  = (half_t*)(ws + wsAlloc(off, (size_t)256 * 512 * 2));
    half_t* W3_h  = (half_t*)(ws + wsAlloc(off, (size_t)128 * 256 * 2));
    half_t* W4_h  = (half_t*)(ws + wsAlloc(off, (size_t)HH * 128 * 2));
    half_t* hs    = (half_t*)(ws + wsAlloc(off, (size_t)BB * TT * HH * 2)); // (B,T,H)
    float*  a1    = (float* )(ws + wsAlloc(off, (size_t)BB * 512 * 4));
    float*  a2    = (float* )(ws + wsAlloc(off, (size_t)BB * 256 * 4));
    float*  a3    = (float* )(ws + wsAlloc(off, (size_t)BB * 128 * 4));
    // Zall last: used only if workspace is big enough
    size_t zoff = wsAlloc(off, (size_t)BB * TT * GG * 2);                   // (B,T,4H) f16
    half_t* Zall = (half_t*)(ws + zoff);
    const int useZall = (ws_size >= off) ? 1 : 0;

    float* out   = (float*)d_out;                  // (B, T, O)
    float* c0out = out + (size_t)BB * TT * OO;     // (B, H) zeros
    float* sgout = c0out + (size_t)BB * HH;        // (B, H)

    auto cvt = [&](const float* src, half_t* dst, int n) {
        cvt_f32_f16_k<<<(n + 255) / 256, 256, 0, stream>>>(src, dst, n);
    };
    cvt(Wih, Wih_h, GG * II);
    cvt(Whh, Whh_h, GG * HH);
    cvt(Wd,  Wd_h,  OO * HH);
    cvt(W1,  W1_h,  512 * OO);
    cvt(W2,  W2_h,  256 * 512);
    cvt(W3,  W3_h,  128 * 256);
    cvt(W4,  W4_h,  HH * 128);

    zero_f32_k<<<(BB * HH + 255) / 256, 256, 0, stream>>>(c0out, BB * HH);

    // ---- hoisted x-part: Zall = X @ Wih^T + b, (8192 x 2048, K=256), f16 out ----
    if (useZall) {
        // M=8192 rows (row = b*T + t matches x row layout), 512*32 tiles -> 2048 blocks
        wmma_gemm_k<false, false, true, true><<<2048, 256, 0, stream>>>(
            x, II, Wih_h, II, b_lstm, Zall, GG, (BB * TT) / 16, GG / 64);
    }

    // ---- persistent fused recurrence: 4 WGs x 512 threads, zero launches in chain ----
    lstm_persistent_k<<<4, 512, 0, stream>>>(x, Zall, Wih_h, Whh_h, b_lstm, hs, useZall);

    // ---- dense readout: out = hs @ Wd^T + bd  (8192x128, K=512) ----
    wmma_gemm_k<true, false, true, false><<<128, 256, 0, stream>>>(
        hs, HH, Wd_h, HH, bd, out, OO, (BB * TT) / 16, OO / 64);

    // ---- synthetic-gradient MLP on out[:, T-1, :] ----
    wmma_gemm_k<false, true, true, false><<<4, 256, 0, stream>>>(
        out + (size_t)(TT - 1) * OO, TT * OO, W1_h, OO, b1, a1, 512, 4, 8);
    wmma_gemm_k<false, true, true, false><<<2, 256, 0, stream>>>(
        a1, 512, W2_h, 512, b2, a2, 256, 4, 4);
    wmma_gemm_k<false, true, true, false><<<1, 256, 0, stream>>>(
        a2, 256, W3_h, 256, b3, a3, 128, 4, 2);
    wmma_gemm_k<false, false, true, false><<<4, 256, 0, stream>>>(
        a3, 128, W4_h, OO, b4, sgout, HH, 4, 8);
}
